// ModSwin2SRSelfAttention_28381143892671
// MI455X (gfx1250) — compile-verified, matrix operand order checked by reading, not run on hardware
//
#include <hip/hip_runtime.h>
#include <hip/hip_bf16.h>
#include <math.h>

// ---- problem constants (from reference) ----
#define BWIN   4096
#define NTOK   64
#define CCH    192
#define NHEAD  6
#define HDIM   32
#define TABLE  225
#define CPBH   512

typedef __attribute__((ext_vector_type(16))) _Float16 v16h;
typedef __attribute__((ext_vector_type(8)))  float    v8f;

// ---------------------------------------------------------------------------
// CDNA5 WMMA fragment loaders (wave32).
// A-matrix 16x32 f16 layout (ISA 7.12.2): lane L holds row M = L%16.
//   lanes 0-15 : VGPR r<4 -> K=2r,2r+1 ; r>=4 -> K=16+2(r-4)+h
//   lanes 16-31: same +8 in each K group
// => k(r,h,laneHi) = 2r + h + (r>=4 ? 8 : 0) + (laneHi ? 8 : 0)
// B-matrix (KxN): lane L holds column N = L%16, same K distribution.
// ---------------------------------------------------------------------------
__device__ __forceinline__ v16h load_frag_h(const _Float16* src, int row0,
                                            int col0, int ld, int lane) {
  v16h f;
  const int m  = row0 + (lane & 15);
  const int kb = (lane & 16) ? 8 : 0;
  const _Float16* p = src + m * ld + col0;
#pragma unroll
  for (int r = 0; r < 8; ++r) {
    const int k = 2 * r + ((r >= 4) ? 8 : 0) + kb;
    f[2 * r]     = p[k];
    f[2 * r + 1] = p[k + 1];
  }
  return f;
}

__device__ __forceinline__ v16h load_frag_h_scaled(const _Float16* src, int row0,
                                                   int col0, int ld, int lane,
                                                   float s) {
  v16h f;
  const int m  = row0 + (lane & 15);
  const int kb = (lane & 16) ? 8 : 0;
  const _Float16* p = src + m * ld + col0;
#pragma unroll
  for (int r = 0; r < 8; ++r) {
    const int k = 2 * r + ((r >= 4) ? 8 : 0) + kb;
    f[2 * r]     = (_Float16)((float)p[k] * s);
    f[2 * r + 1] = (_Float16)((float)p[k + 1] * s);
  }
  return f;
}

// B-fragment for D = X * W^T : B[k][n] = W[n][k], W row-major f32 in global.
__device__ __forceinline__ v16h load_frag_w(const float* src, int row0,
                                            int col0, int ld, int lane) {
  v16h f;
  const int n  = row0 + (lane & 15);
  const int kb = (lane & 16) ? 8 : 0;
  const float* p = src + n * ld + col0;
#pragma unroll
  for (int r = 0; r < 8; ++r) {
    const int k = 2 * r + ((r >= 4) ? 8 : 0) + kb;
    f[2 * r]     = (_Float16)p[k];
    f[2 * r + 1] = (_Float16)p[k + 1];
  }
  return f;
}

// B-fragment for truly row-major KxN matrix (used for probs @ V).
__device__ __forceinline__ v16h load_frag_col(const _Float16* src, int k0,
                                              int n0, int ld, int lane) {
  v16h f;
  const int n  = n0 + (lane & 15);
  const int kb = (lane & 16) ? 8 : 0;
#pragma unroll
  for (int r = 0; r < 8; ++r) {
    const int k = k0 + 2 * r + ((r >= 4) ? 8 : 0) + kb;
    f[2 * r]     = src[k * ld + n];
    f[2 * r + 1] = src[(k + 1) * ld + n];
  }
  return f;
}

// ---------------------------------------------------------------------------
// Kernel A: continuous position bias MLP, 16*sigmoid folded in.
// out[t*6+h] = 16*sigmoid( sum_c relu(coords[t]·w1[c] + b1[c]) * w2[h][c] )
// ---------------------------------------------------------------------------
__global__ void cpb_table_kernel(const float* __restrict__ w1,
                                 const float* __restrict__ b1,
                                 const float* __restrict__ w2,
                                 const float* __restrict__ rct,
                                 float* __restrict__ outb) {
  const int e = blockIdx.x * blockDim.x + threadIdx.x;
  if (e >= TABLE * NHEAD) return;
  const int t = e / NHEAD, h = e % NHEAD;
  const float x = rct[t * 2 + 0], y = rct[t * 2 + 1];
  float acc = 0.f;
  for (int c = 0; c < CPBH; ++c) {
    const float hid = fmaxf(x * w1[c * 2] + y * w1[c * 2 + 1] + b1[c], 0.f);
    acc += hid * w2[h * CPBH + c];
  }
  outb[e] = 16.f / (1.f + expf(-acc));
}

// ---------------------------------------------------------------------------
// Kernel B: fully fused Swin2SR window attention. One block = one window.
// 8 waves, ~99KB dynamic LDS (sX aliases sS/sP: phase-disjoint).
// ---------------------------------------------------------------------------
__global__ void __launch_bounds__(256)
swin2sr_attn_kernel(const float* __restrict__ hs,
                    const float* __restrict__ mask,
                    const float* __restrict__ qw, const float* __restrict__ qb,
                    const float* __restrict__ kw, const float* __restrict__ kbias,
                    const float* __restrict__ vw, const float* __restrict__ vb,
                    const float* __restrict__ lsc,
                    const int*   __restrict__ rpi,
                    const float* __restrict__ wsb,   // [225][6] 16*sigmoid(bias)
                    float* __restrict__ out) {
  extern __shared__ char smem[];
  float*    sS   = (float*)smem;                      // 64*64 f32   (16384 B)
  _Float16* sP   = (_Float16*)(smem + 16384);         // 64*64 f16   ( 8192 B)
  _Float16* sX   = (_Float16*)smem;                   // 64*192 f16, aliases sS+sP
  _Float16* sQ   = (_Float16*)(smem + 24576);         // 64*192 f16
  _Float16* sK   = (_Float16*)(smem + 49152);
  _Float16* sV   = (_Float16*)(smem + 73728);
  float*    sQrn = (float*)(smem + 98304);            // [6][64]
  float*    sKrn = (float*)(smem + 99840);            // [6][64]

  const int tid  = threadIdx.x;
  const int lane = tid & 31;
  const int wid  = tid >> 5;
  const int blk  = blockIdx.x;

  // warm L2 with the (shared, tiny) projection weights: global_prefetch_b8
  if (blk < 64) {
    for (int r = tid; r < CCH; r += 256) {
      __builtin_prefetch(qw + r * CCH, 0, 3);
      __builtin_prefetch(kw + r * CCH, 0, 3);
      __builtin_prefetch(vw + r * CCH, 0, 3);
    }
  }

  // ---- phase 1: X[64][192] f32 -> f16 LDS (vectorized b128 loads) ----
  {
    const float4* src = (const float4*)(hs + (size_t)blk * NTOK * CCH);
    for (int e = tid; e < NTOK * CCH / 4; e += 256) {
      const float4 x = src[e];
      const int o = e * 4;
      sX[o + 0] = (_Float16)x.x;
      sX[o + 1] = (_Float16)x.y;
      sX[o + 2] = (_Float16)x.z;
      sX[o + 3] = (_Float16)x.w;
    }
  }
  __syncthreads();

  // ---- phase 2: Q/K/V = X @ W^T + b via WMMA (144 tiles, K=192) ----
  for (int t = wid; t < 144; t += 8) {
    const int proj = t / 48, rem = t % 48, mt = rem / 12, nt = rem % 12;
    const float* Wm  = (proj == 0) ? qw : ((proj == 1) ? kw : vw);
    const float* Bm  = (proj == 0) ? qb : ((proj == 1) ? kbias : vb);
    _Float16*    dst = (proj == 0) ? sQ : ((proj == 1) ? sK : sV);
    v8f acc = {};
#pragma unroll
    for (int kk = 0; kk < 6; ++kk) {
      const v16h a = load_frag_h(sX, mt * 16, kk * 32, CCH, lane);
      const v16h b = load_frag_w(Wm, nt * 16, kk * 32, CCH, lane);
      acc = __builtin_amdgcn_wmma_f32_16x16x32_f16(false, a, false, b,
                                                   (short)0, acc, false, false);
    }
    const int n  = nt * 16 + (lane & 15);
    const float bv = Bm[n];
#pragma unroll
    for (int r = 0; r < 8; ++r) {
      const int m = mt * 16 + r + ((lane & 16) ? 8 : 0);
      dst[m * CCH + n] = (_Float16)(acc[r] + bv);
    }
  }
  __syncthreads();

  // ---- phase 3: cosine-attention row norms (1/max(||.||, 1e-12)) ----
  for (int e = tid; e < 2 * NHEAD * NTOK; e += 256) {
    const int qk = e >= NHEAD * NTOK;
    const int ee = e - qk * NHEAD * NTOK;
    const int h = ee >> 6, m = ee & 63;
    const _Float16* src = qk ? sK : sQ;
    float s = 0.f;
#pragma unroll
    for (int d = 0; d < HDIM; ++d) {
      const float x = (float)src[m * CCH + h * HDIM + d];
      s += x * x;
    }
    (qk ? sKrn : sQrn)[ee] = 1.f / fmaxf(sqrtf(s), 1e-12f);
  }
  __syncthreads();

  // ---- phase 4: per-head attention ----
  for (int h = 0; h < NHEAD; ++h) {
    const float hscale = expf(fminf(lsc[h], 4.6051702f));  // log(100)

    // scores = hscale * qn @ kn^T + bias16 + 2*mask   (16 tiles / 8 waves)
    for (int t = wid; t < 16; t += 8) {
      const int mt = t >> 2, nt = t & 3;
      const float qs = sQrn[h * NTOK + mt * 16 + (lane & 15)];
      const float ks = sKrn[h * NTOK + nt * 16 + (lane & 15)];
      const v16h a = load_frag_h_scaled(sQ, mt * 16, h * HDIM, CCH, lane, qs);
      const v16h b = load_frag_h_scaled(sK, nt * 16, h * HDIM, CCH, lane, ks);
      v8f c = {};
      c = __builtin_amdgcn_wmma_f32_16x16x32_f16(false, a, false, b,
                                                 (short)0, c, false, false);
      const int j = nt * 16 + (lane & 15);
#pragma unroll
      for (int r = 0; r < 8; ++r) {
        const int i   = mt * 16 + r + ((lane & 16) ? 8 : 0);
        const int idx = rpi[i * NTOK + j];
        const float bv = wsb[idx * NHEAD + h];
        const float mv = mask[((size_t)blk * NTOK + i) * NTOK + j];
        sS[i * NTOK + j] = c[r] * hscale + bv + 2.f * mv;  // mask added twice
      }
    }
    __syncthreads();

    // softmax over rows of 64: 4 lanes/row, shfl_xor reductions (wave32)
    {
      const int row = tid >> 2, part = tid & 3;
      float v[16];
      float mx = -3.4e38f;
      const float* srow = sS + row * NTOK + part * 16;
#pragma unroll
      for (int i = 0; i < 16; ++i) { v[i] = srow[i]; mx = fmaxf(mx, v[i]); }
      mx = fmaxf(mx, __shfl_xor(mx, 1, 32));
      mx = fmaxf(mx, __shfl_xor(mx, 2, 32));
      float sm = 0.f;
#pragma unroll
      for (int i = 0; i < 16; ++i) { v[i] = expf(v[i] - mx); sm += v[i]; }
      sm += __shfl_xor(sm, 1, 32);
      sm += __shfl_xor(sm, 2, 32);
      const float inv = 1.f / sm;
      _Float16* prow = sP + row * NTOK + part * 16;
#pragma unroll
      for (int i = 0; i < 16; ++i) prow[i] = (_Float16)(v[i] * inv);
    }
    __syncthreads();

    // ctx = probs @ V_head  (8 tiles = 8 waves; K=64 -> 2 WMMA)
    {
      const int mt = wid >> 1, nt = wid & 1;
      v8f c = {};
#pragma unroll
      for (int kk = 0; kk < 2; ++kk) {
        const v16h a = load_frag_h(sP, mt * 16, kk * 32, NTOK, lane);
        const v16h b = load_frag_col(sV, kk * 32, h * HDIM + nt * 16, CCH, lane);
        c = __builtin_amdgcn_wmma_f32_16x16x32_f16(false, a, false, b,
                                                   (short)0, c, false, false);
      }
      const int n = h * HDIM + nt * 16 + (lane & 15);
#pragma unroll
      for (int r = 0; r < 8; ++r) {
        const int m = mt * 16 + r + ((lane & 16) ? 8 : 0);
        out[((size_t)blk * NTOK + m) * CCH + n] = c[r];
      }
    }
    __syncthreads();
  }
}

extern "C" void kernel_launch(void* const* d_in, const int* in_sizes, int n_in,
                              void* d_out, int out_size, void* d_ws, size_t ws_size,
                              hipStream_t stream) {
  (void)in_sizes; (void)n_in; (void)out_size; (void)ws_size;
  const float* hs    = (const float*)d_in[0];
  const float* mask  = (const float*)d_in[1];
  const float* qw    = (const float*)d_in[2];
  const float* qb    = (const float*)d_in[3];
  const float* kw    = (const float*)d_in[4];
  const float* kbias = (const float*)d_in[5];
  const float* vw    = (const float*)d_in[6];
  const float* vb    = (const float*)d_in[7];
  const float* lsc   = (const float*)d_in[8];
  const float* w1    = (const float*)d_in[9];
  const float* b1    = (const float*)d_in[10];
  const float* w2    = (const float*)d_in[11];
  const float* rct   = (const float*)d_in[12];
  const int*   rpi   = (const int*)d_in[13];
  float* out = (float*)d_out;
  float* wsb = (float*)d_ws;  // 225*6 floats

  cpb_table_kernel<<<(TABLE * NHEAD + 255) / 256, 256, 0, stream>>>(
      w1, b1, w2, rct, wsb);

  const size_t lds_bytes = 101376;  // 99 KB: 3 workgroups / 320KB WGP
  swin2sr_attn_kernel<<<BWIN, 256, lds_bytes, stream>>>(
      hs, mask, qw, qb, kw, kbias, vw, vb, lsc, rpi, wsb, out);
}
